// PNP_19954418057708
// MI455X (gfx1250) — matrix operand order repaired and checked by reading, not compiled, not run on hardware
//
#include <hip/hip_runtime.h>
#include <math.h>

// ---------------------------------------------------------------------------
// CLIP-vocab prototype mixture weights, MI455X (gfx1250, wave32, WMMA f16).
// Big GEMM (8192x30000x768) fused with per-batch patch-max; logits never hit
// memory. All reductions are fixed-order (deterministic). ws ~185 MB.
// ---------------------------------------------------------------------------

typedef _Float16 f16;
typedef __attribute__((ext_vector_type(8)))  f16   v8h;
typedef __attribute__((ext_vector_type(16))) f16   v16h;
typedef __attribute__((ext_vector_type(8)))  float v8f;

union AFrag { v16h v; v8h h[2]; };

#define NB    8
#define NP    1024
#define DD    768
#define VV    30000
#define VPAD  30080          /* 235 * 128 */
#define CC    512
#define HH    768
#define INVT  5.0f           /* 1 / 0.2 */

// ---------- row L2-normalize f32 -> f16 (zero pad rows) --------------------
__global__ void pnp_norm_rows(const float* __restrict__ src, f16* __restrict__ dst,
                              int rows, int cols) {
  const int row = blockIdx.x, tid = threadIdx.x;
  __shared__ float red[256];
  if (row >= rows) {                       // uniform branch per block
    for (int i = tid; i < cols; i += 256) dst[(size_t)row * cols + i] = (f16)0.0f;
    return;
  }
  const float* s = src + (size_t)row * cols;
  float ss = 0.f;
  for (int i = tid; i < cols; i += 256) { float v = s[i]; ss += v * v; }
  red[tid] = ss; __syncthreads();
  for (int k = 128; k > 0; k >>= 1) { if (tid < k) red[tid] += red[tid + k]; __syncthreads(); }
  const float inv = 1.0f / fmaxf(sqrtf(red[0]), 1e-12f);
  for (int i = tid; i < cols; i += 256) dst[(size_t)row * cols + i] = (f16)(s[i] * inv);
}

// ---------- transpose f32 [R][C] -> f16 [C][R] -----------------------------
__global__ void pnp_transpose_f16(const float* __restrict__ src, f16* __restrict__ dst,
                                  int R, int C) {
  int idx = blockIdx.x * 256 + threadIdx.x;
  if (idx >= R * C) return;
  int r = idx / C, c = idx - r * C;
  dst[(size_t)c * R + r] = (f16)src[idx];
}

// ---------- GEMM1: h[VPAD][768] = vocab16[VPAD][512] @ W1t ----------------
__global__ void pnp_gemm1(const f16* __restrict__ A, const f16* __restrict__ Bt,
                          float* __restrict__ H) {
  const int tid = threadIdx.x, wave = tid >> 5, lane = tid & 31;
  const int hi = lane >> 4, lm = lane & 15;
  const int row0 = blockIdx.y * 128 + wave * 16;
  const int col0 = blockIdx.x * 16;
  v8f acc = {};
  const f16* arow = A  + (size_t)(row0 + lm) * CC;
  const f16* brow = Bt + (size_t)(col0 + lm) * CC;
  for (int k0 = 0; k0 < CC; k0 += 32) {
    AFrag a;
    a.h[0] = *(const v8h*)(arow + k0 + hi * 16);
    a.h[1] = *(const v8h*)(arow + k0 + hi * 16 + 8);
    v16h bf = *(const v16h*)(brow + k0 + hi * 16);
    acc = __builtin_amdgcn_wmma_f32_16x16x32_f16(false, a.v, false, bf,
                                                 (short)0, acc, false, false);
  }
  float* hp = H + (size_t)row0 * HH + col0 + lm;
  for (int r = 0; r < 8; ++r) hp[(size_t)(r + 8 * hi) * HH] = acc[r];
}

// ---------- per-column BN stats over V rows (fixed-order) ------------------
__global__ void pnp_colstats(const float* __restrict__ H, const float* __restrict__ gamma,
                             float* __restrict__ meanO, float* __restrict__ scaleO) {
  const int c = blockIdx.x, tid = threadIdx.x;
  __shared__ float rs[256], rq[256];
  float s = 0.f, q = 0.f;
  for (int v = tid; v < VV; v += 256) { float h = H[(size_t)v * HH + c]; s += h; q += h * h; }
  rs[tid] = s; rq[tid] = q; __syncthreads();
  for (int k = 128; k > 0; k >>= 1) {
    if (tid < k) { rs[tid] += rs[tid + k]; rq[tid] += rq[tid + k]; }
    __syncthreads();
  }
  if (tid == 0) {
    float m   = rs[0] / (float)VV;
    float var = rq[0] / (float)VV - m * m;       // biased, like torch BN
    meanO[c]  = m;
    scaleO[c] = gamma[c] * rsqrtf(var + 1e-5f);
  }
}

// ---------- GEMM2: protos = l2norm(relu(bn(h)) @ W2t + b2) -> f16 ----------
// one block = 16 rows x all 768 cols (8 waves x 6 N-tiles), norm fused in LDS
__global__ void pnp_gemm2(const float* __restrict__ H, const float* __restrict__ mean,
                          const float* __restrict__ scale, const float* __restrict__ beta,
                          const f16* __restrict__ W2t, const float* __restrict__ b2,
                          f16* __restrict__ P) {
  const int tid = threadIdx.x, wave = tid >> 5, lane = tid & 31;
  const int hi = lane >> 4, lm = lane & 15;
  const int row0 = blockIdx.x * 16;
  __shared__ float red[8][16];
  __shared__ float invn[16];
  v8f acc[6] = {};
  const float* hrow = H + (size_t)(row0 + lm) * HH;
  for (int k0 = 0; k0 < HH; k0 += 32) {
    AFrag a;
    #pragma unroll
    for (int g = 0; g < 2; ++g) {
      const int kb = k0 + hi * 16 + g * 8;
      #pragma unroll
      for (int i = 0; i < 8; ++i) {
        float x = (hrow[kb + i] - mean[kb + i]) * scale[kb + i] + beta[kb + i];
        a.h[g][i] = (f16)fmaxf(x, 0.0f);
      }
    }
    #pragma unroll
    for (int j = 0; j < 6; ++j) {
      const int col0 = (wave * 6 + j) * 16;
      v16h bf = *(const v16h*)(W2t + (size_t)(col0 + lm) * HH + k0 + hi * 16);
      acc[j] = __builtin_amdgcn_wmma_f32_16x16x32_f16(false, a.v, false, bf,
                                                      (short)0, acc[j], false, false);
    }
  }
  float rowsq[8];
  #pragma unroll
  for (int r = 0; r < 8; ++r) rowsq[r] = 0.f;
  #pragma unroll
  for (int j = 0; j < 6; ++j) {
    const float bb = b2[(wave * 6 + j) * 16 + lm];
    #pragma unroll
    for (int r = 0; r < 8; ++r) {
      acc[j][r] = acc[j][r] + bb;
      rowsq[r] += acc[j][r] * acc[j][r];
    }
  }
  #pragma unroll
  for (int r = 0; r < 8; ++r)
    for (int m = 1; m < 16; m <<= 1)
      rowsq[r] += __shfl_xor(rowsq[r], m, 32);   // reduce 16 lanes of this half
  if (lm == 0)
    for (int r = 0; r < 8; ++r) red[wave][r + 8 * hi] = rowsq[r];
  __syncthreads();
  if (tid < 16) {
    float s = 0.f;
    for (int w = 0; w < 8; ++w) s += red[w][tid]; // fixed order across waves
    invn[tid] = 1.0f / fmaxf(sqrtf(s), 1e-12f);
  }
  __syncthreads();
  #pragma unroll
  for (int j = 0; j < 6; ++j) {
    const int col = (wave * 6 + j) * 16 + lm;
    #pragma unroll
    for (int r = 0; r < 8; ++r)
      P[(size_t)(row0 + r + 8 * hi) * HH + col] = (f16)(acc[j][r] * invn[r + 8 * hi]);
  }
}

// ---------- GEMM3: fused logits + max over the 1024 patches ----------------
// block = 128 vocab cols of one batch; each wave owns 16 cols, loops all M.
__global__ void pnp_gemm3(const f16* __restrict__ PT, const f16* __restrict__ P,
                          float* __restrict__ VL) {
  const int tid = threadIdx.x, wave = tid >> 5, lane = tid & 31;
  const int hi = lane >> 4, lm = lane & 15;
  const int b = blockIdx.y;
  const int colv = blockIdx.x * 128 + wave * 16 + lm;       // < VPAD always
  const f16* brow  = P  + (size_t)colv * DD;
  const f16* abase = PT + (size_t)b * NP * DD;
  float runmax = -3.4e38f;
  for (int m0 = 0; m0 < NP; m0 += 64) {
    if (m0 + 64 < NP)                                        // hint next A tile
      __builtin_prefetch(abase + (size_t)(m0 + 64 + lm) * DD, 0, 0);
    v8f acc[4] = {};
    for (int k0 = 0; k0 < DD; k0 += 32) {
      v16h bf = *(const v16h*)(brow + k0 + hi * 16);          // shared by 4 tiles
      #pragma unroll
      for (int s = 0; s < 4; ++s) {
        const f16* arow = abase + (size_t)(m0 + s * 16 + lm) * DD;
        AFrag a;
        a.h[0] = *(const v8h*)(arow + k0 + hi * 16);
        a.h[1] = *(const v8h*)(arow + k0 + hi * 16 + 8);
        acc[s] = __builtin_amdgcn_wmma_f32_16x16x32_f16(false, a.v, false, bf,
                                                        (short)0, acc[s], false, false);
      }
    }
    #pragma unroll
    for (int s = 0; s < 4; ++s) {                            // max over 16 rows
      float tm = acc[s][0];
      #pragma unroll
      for (int r = 1; r < 8; ++r) tm = fmaxf(tm, acc[s][r]);
      tm = fmaxf(tm, __shfl_xor(tm, 16, 32));                // other 8 rows
      runmax = fmaxf(runmax, tm);
    }
  }
  if (hi == 0 && colv < VV) VL[b * VV + colv] = runmax;
}

// ---------- softmax(vocab_logits / T) per batch ----------------------------
__global__ void pnp_softmax(const float* __restrict__ VL, float* __restrict__ out) {
  const int b = blockIdx.x, tid = threadIdx.x;
  __shared__ float red[256];
  const float* row = VL + b * VV;
  float mx = -3.4e38f;
  for (int i = tid; i < VV; i += 256) mx = fmaxf(mx, row[i]);
  red[tid] = mx; __syncthreads();
  for (int k = 128; k > 0; k >>= 1) { if (tid < k) red[tid] = fmaxf(red[tid], red[tid + k]); __syncthreads(); }
  mx = red[0]; __syncthreads();
  float sum = 0.f;
  for (int i = tid; i < VV; i += 256) sum += expf((row[i] - mx) * INVT);
  red[tid] = sum; __syncthreads();
  for (int k = 128; k > 0; k >>= 1) { if (tid < k) red[tid] += red[tid + k]; __syncthreads(); }
  const float inv = 1.0f / red[0];
  for (int i = tid; i < VV; i += 256) out[b * VV + i] = expf((row[i] - mx) * INVT) * inv;
}

// ---------------------------------------------------------------------------
extern "C" void kernel_launch(void* const* d_in, const int* in_sizes, int n_in,
                              void* d_out, int out_size, void* d_ws, size_t ws_size,
                              hipStream_t stream) {
  const float* pt    = (const float*)d_in[0];   // [8,1024,768]
  const float* vocab = (const float*)d_in[1];   // [30000,512]
  const float* W1    = (const float*)d_in[2];   // [512,768]
  const float* gamma = (const float*)d_in[3];   // [768]
  const float* beta  = (const float*)d_in[4];   // [768]
  const float* W2    = (const float*)d_in[5];   // [768,768]
  const float* b2    = (const float*)d_in[6];   // [768]
  float* out = (float*)d_out;                   // [8,30000]

  char* w = (char*)d_ws;
  auto alloc = [&](size_t bytes) -> char* {
    char* p = w; w += (bytes + 255) & ~(size_t)255; return p;
  };
  f16*   pt16   = (f16*)  alloc((size_t)NB * NP * DD * 2);    // 12.6 MB
  f16*   voc16  = (f16*)  alloc((size_t)VPAD * CC * 2);       // 30.8 MB
  f16*   W1t    = (f16*)  alloc((size_t)HH * CC * 2);         //  0.8 MB
  f16*   W2t    = (f16*)  alloc((size_t)DD * HH * 2);         //  1.2 MB
  float* H      = (float*)alloc((size_t)VPAD * HH * 4);       // 92.4 MB
  float* meanp  = (float*)alloc((size_t)HH * 4);
  float* scalep = (float*)alloc((size_t)HH * 4);
  f16*   P16    = (f16*)  alloc((size_t)VPAD * DD * 2);       // 46.2 MB
  float* VL     = (float*)alloc((size_t)NB * VV * 4);         //  1.0 MB

  pnp_norm_rows<<<NB * NP, 256, 0, stream>>>(pt, pt16, NB * NP, DD);
  pnp_norm_rows<<<VPAD,    256, 0, stream>>>(vocab, voc16, VV, CC);
  pnp_transpose_f16<<<(CC * HH + 255) / 256, 256, 0, stream>>>(W1, W1t, CC, HH);
  pnp_transpose_f16<<<(HH * DD + 255) / 256, 256, 0, stream>>>(W2, W2t, HH, DD);
  pnp_gemm1<<<dim3(HH / 16, VPAD / 128), 256, 0, stream>>>(voc16, W1t, H);
  pnp_colstats<<<HH, 256, 0, stream>>>(H, gamma, meanp, scalep);
  pnp_gemm2<<<VPAD / 16, 256, 0, stream>>>(H, meanp, scalep, beta, W2t, b2, P16);
  pnp_gemm3<<<dim3(VPAD / 128, NB), 256, 0, stream>>>(pt16, P16, VL);
  pnp_softmax<<<NB, 256, 0, stream>>>(VL, out);
}
